// Drugcell_38981123178855
// MI455X (gfx1250) — compile-verified, matrix operand order checked by reading, not run on hardware
//
#include <hip/hip_runtime.h>
#include <hip/hip_bf16.h>
#include <math.h>

#define GENE_K   3008
#define GENE_N   16384   // 2048 terms * 8
#define BATCH    4096
#define NTERM0   2048
#define HH       6
#define BK       32
#define BM       128
#define BN       128
#define EPSV     1e-5f

typedef __attribute__((ext_vector_type(16))) __bf16 v16bf;
typedef __attribute__((ext_vector_type(8)))  __bf16 v8bf;
typedef __attribute__((ext_vector_type(4)))  __bf16 v4bf;
typedef __attribute__((ext_vector_type(8)))  float  v8f;

union V16 { v16bf v; v8bf h[2]; };

// ---------------------------------------------------------------------------
// Fused gene-GEMM (bf16 WMMA, f32 accum) + layer-0 (per-term 8->6, bias, tanh)
// Writes h0_raw (B x 2048 x 6) f32 (pre-batchnorm).
// grid = (BATCH/BM, GENE_N/BN), block = 256 (8 waves, 4(M) x 2(N) wave grid,
// wave tile 32x64 -> 2x4 wmma tiles = 8 v_wmma per wave per K-step).
// ---------------------------------------------------------------------------
__global__ __launch_bounds__(256)
void gene_gemm_l0(const float* __restrict__ x,   // (B, 3008)
                  const float* __restrict__ Wg,  // (16384, 3008) == (2048,8,3008)
                  const float* __restrict__ bg,  // (2048, 8)
                  const float* __restrict__ W0,  // (2048, 6, 8)
                  const float* __restrict__ b0,  // (2048, 6)
                  float* __restrict__ h0)        // (B, 2048, 6)
{
    __shared__ __align__(16) __bf16 As[BM * BK];   // x tile, row-major (m,k)   8KB
    __shared__ __align__(16) __bf16 Bs[BN * BK];   // Wg tile, row-major (n,k)  8KB
    __shared__ float Cs[BM * BN];                  // 64KB

    const int tid      = threadIdx.x;
    const int lane     = tid & 31;
    const int wave     = tid >> 5;
    const int waveM    = (wave & 3) * 32;    // 4 waves along M
    const int waveN    = (wave >> 2) * 64;   // 2 waves along N
    const int laneMod  = lane & 15;
    const int laneHalf = lane >> 4;          // 0 | 1
    const int kSel     = laneHalf * 8;       // ISA 16-bit A/B lane K-split

    const int brow0 = blockIdx.x * BM;       // batch rows (fast grid dim -> Wg L2 reuse)
    const int n0    = blockIdx.y * BN;       // output columns (16 terms)

    v8f acc[2][4] = {};

    for (int k0 = 0; k0 < GENE_K; k0 += BK) {
        // ---- stage tiles to LDS with on-the-fly f32 -> bf16 conversion ----
#pragma unroll
        for (int i = 0; i < 4; ++i) {
            const int id = tid + i * 256;    // 0..1023 : 128 rows x 8 float4-chunks
            const int r  = id >> 3;
            const int c4 = (id & 7) * 4;
            const float4 fx = *(const float4*)(x  + (size_t)(brow0 + r) * GENE_K + k0 + c4);
            const float4 fw = *(const float4*)(Wg + (size_t)(n0    + r) * GENE_K + k0 + c4);
            v4bf va, vb;
            va[0] = (__bf16)fx.x; va[1] = (__bf16)fx.y; va[2] = (__bf16)fx.z; va[3] = (__bf16)fx.w;
            vb[0] = (__bf16)fw.x; vb[1] = (__bf16)fw.y; vb[2] = (__bf16)fw.z; vb[3] = (__bf16)fw.w;
            *(v4bf*)&As[r * BK + c4] = va;
            *(v4bf*)&Bs[r * BK + c4] = vb;
            if (k0 + BK < GENE_K) {          // gfx1250 global_prefetch of next K tile
                __builtin_prefetch(x  + (size_t)(brow0 + r) * GENE_K + k0 + BK + c4, 0, 3);
                __builtin_prefetch(Wg + (size_t)(n0    + r) * GENE_K + k0 + BK + c4, 0, 3);
            }
        }
        __syncthreads();

        // ---- build fragments (ISA 16-bit A layout; B staged as N x K) ----
        V16 af[2], bf2[4];
#pragma unroll
        for (int i = 0; i < 2; ++i) {
            const __bf16* pa = &As[(waveM + i * 16 + laneMod) * BK + kSel];
            af[i].h[0] = *(const v8bf*)pa;
            af[i].h[1] = *(const v8bf*)(pa + 16);
        }
#pragma unroll
        for (int j = 0; j < 4; ++j) {
            const __bf16* pb = &Bs[(waveN + j * 16 + laneMod) * BK + kSel];
            bf2[j].h[0] = *(const v8bf*)pb;
            bf2[j].h[1] = *(const v8bf*)(pb + 16);
        }
#pragma unroll
        for (int i = 0; i < 2; ++i)
#pragma unroll
            for (int j = 0; j < 4; ++j)
                acc[i][j] = __builtin_amdgcn_wmma_f32_16x16x32_bf16(
                    false, af[i].v, false, bf2[j].v, (short)0, acc[i][j], false, false);
        __syncthreads();
    }

    // ---- spill C to LDS (ISA C/D layout: vgpr r -> M=r(+8), lane -> N) ----
#pragma unroll
    for (int i = 0; i < 2; ++i)
#pragma unroll
        for (int j = 0; j < 4; ++j)
#pragma unroll
            for (int r = 0; r < 8; ++r) {
                const int row = waveM + i * 16 + laneHalf * 8 + r;
                const int col = waveN + j * 16 + laneMod;
                Cs[row * BN + col] = acc[i][j][r];
            }
    __syncthreads();

    // ---- fused layer-0: per (row, term) 8->6 matmul + bias + tanh ----
    for (int p = tid; p < BM * (BN / 8); p += 256) {
        const int row = p >> 4;              // 0..127
        const int tl  = p & 15;              // 0..15 local term
        const int t   = (n0 >> 3) + tl;      // global term id
        float gv[8];
#pragma unroll
        for (int o = 0; o < 8; ++o)
            gv[o] = Cs[row * BN + tl * 8 + o] + bg[t * 8 + o];
        float* __restrict__ orow = h0 + (size_t)(brow0 + row) * (NTERM0 * HH) + t * HH;
#pragma unroll
        for (int op = 0; op < HH; ++op) {
            float a = b0[t * HH + op];
#pragma unroll
            for (int o = 0; o < 8; ++o)
                a += gv[o] * W0[t * 48 + op * 8 + o];
            orow[op] = tanhf(a);
        }
    }
}

// ---------------------------------------------------------------------------
// Per-feature batch statistics: mu[f], rstd[f] over BATCH rows of h (B x F).
// grid.x = ceil(F/64), block = 256 (64 features x 4 batch-partials)
// ---------------------------------------------------------------------------
__global__ __launch_bounds__(256)
void stats_kernel(const float* __restrict__ h, int F,
                  float* __restrict__ mu, float* __restrict__ rstd)
{
    __shared__ float ssum[256], ssq[256];
    const int f = blockIdx.x * 64 + (threadIdx.x & 63);
    const int g = threadIdx.x >> 6;          // 0..3
    float s = 0.f, q = 0.f;
    if (f < F) {
        for (int b = g; b < BATCH; b += 4) {
            const float v = h[(size_t)b * F + f];
            s += v; q += v * v;
        }
    }
    ssum[threadIdx.x] = s; ssq[threadIdx.x] = q;
    __syncthreads();
    if (g == 0 && f < F) {
        s = ssum[threadIdx.x] + ssum[threadIdx.x + 64] + ssum[threadIdx.x + 128] + ssum[threadIdx.x + 192];
        q = ssq [threadIdx.x] + ssq [threadIdx.x + 64] + ssq [threadIdx.x + 128] + ssq [threadIdx.x + 192];
        const float m = s * (1.0f / BATCH);
        const float v = q * (1.0f / BATCH) - m * m;
        mu[f]   = m;
        rstd[f] = rsqrtf(v + EPSV);
    }
}

// ---------------------------------------------------------------------------
// Generic term layer i (i>=1): normalize previous layer on-the-fly, micro
// matmul (cin -> 6), bias, tanh.  One thread per (batch, out-term).
// ---------------------------------------------------------------------------
__global__ __launch_bounds__(256)
void term_layer(const float* __restrict__ hin,  // (B, Fin) raw (pre-norm)
                const float* __restrict__ mu, const float* __restrict__ rstd,
                const float* __restrict__ gm, const float* __restrict__ bt,
                const float* __restrict__ W,   // (Tout, 6, cin)
                const float* __restrict__ bias,// (Tout, 6)
                float* __restrict__ hout,      // (B, Tout*6)
                int Tout, int cin, int Fin)
{
    const int idx = blockIdx.x * blockDim.x + threadIdx.x;
    if (idx >= BATCH * Tout) return;
    const int b = idx / Tout;
    const int t = idx % Tout;

    float acc[HH];
#pragma unroll
    for (int o = 0; o < HH; ++o) acc[o] = bias[t * HH + o];

    const float* __restrict__ irow = hin + (size_t)b * Fin + t * cin;
    for (int c = 0; c < cin; ++c) {
        const int fp = t * cin + c;
        const float v = (irow[c] - mu[fp]) * rstd[fp] * gm[fp] + bt[fp];
#pragma unroll
        for (int o = 0; o < HH; ++o)
            acc[o] += v * W[(t * HH + o) * cin + c];
    }
    float* __restrict__ orow = hout + (size_t)b * (Tout * HH) + t * HH;
#pragma unroll
    for (int o = 0; o < HH; ++o) orow[o] = tanhf(acc[o]);
}

// ---------------------------------------------------------------------------
// Final projection: out[b,d] = sum_f hat(root)[b,f] * Wf[d,f] + bf[d]
// ---------------------------------------------------------------------------
__global__ __launch_bounds__(256)
void final_proj(const float* __restrict__ h5,  // (B, 6) raw
                const float* __restrict__ mu, const float* __restrict__ rstd,
                const float* __restrict__ gm, const float* __restrict__ bt,
                const float* __restrict__ Wf,  // (768, 6)
                const float* __restrict__ bfv, // (768,)
                float* __restrict__ out)       // (B, 768)
{
    const int idx = blockIdx.x * blockDim.x + threadIdx.x;
    if (idx >= BATCH * 768) return;
    const int b = idx / 768;
    const int d = idx % 768;
    float a = bfv[d];
#pragma unroll
    for (int f = 0; f < HH; ++f) {
        const float v = (h5[b * HH + f] - mu[f]) * rstd[f] * gm[f] + bt[f];
        a += v * Wf[d * HH + f];
    }
    out[idx] = a;
}

// ---------------------------------------------------------------------------
extern "C" void kernel_launch(void* const* d_in, const int* in_sizes, int n_in,
                              void* d_out, int out_size, void* d_ws, size_t ws_size,
                              hipStream_t stream)
{
    (void)in_sizes; (void)n_in; (void)out_size; (void)ws_size;

    const float* x  = (const float*)d_in[0];
    const float* Wg = (const float*)d_in[1];
    const float* bg = (const float*)d_in[2];
    // layer i params at d_in[3 + 4*i .. 6 + 4*i] : W, b, gm, bt
    const float* W[6], *bb[6], *gm[6], *bt[6];
    for (int i = 0; i < 6; ++i) {
        W [i] = (const float*)d_in[3 + 4 * i + 0];
        bb[i] = (const float*)d_in[3 + 4 * i + 1];
        gm[i] = (const float*)d_in[3 + 4 * i + 2];
        bt[i] = (const float*)d_in[3 + 4 * i + 3];
    }
    const float* Wf  = (const float*)d_in[27];
    const float* bfv = (const float*)d_in[28];
    float* out = (float*)d_out;

    // ---- workspace layout (floats) ----
    const int   Fs[6] = { 12288, 3072, 768, 192, 48, 6 };   // T*6 per layer
    const int   Ts[6] = { 2048, 512, 128, 32, 8, 1 };
    const int   Ci[6] = { 8, 24, 24, 24, 24, 48 };
    float* ws = (float*)d_ws;
    size_t off = 0;
    float* h[6];
    for (int i = 0; i < 6; ++i) { h[i] = ws + off; off += (size_t)BATCH * Fs[i]; }
    float* muv[6]; float* rsd[6];
    for (int i = 0; i < 6; ++i) { muv[i] = ws + off; off += Fs[i];
                                  rsd[i] = ws + off; off += Fs[i]; }

    // ---- 1. fused gene GEMM + layer 0 ----
    {
        dim3 grid(BATCH / BM, GENE_N / BN);
        gene_gemm_l0<<<grid, 256, 0, stream>>>(x, Wg, bg, W[0], bb[0], h[0]);
    }
    stats_kernel<<<dim3((Fs[0] + 63) / 64), 256, 0, stream>>>(h[0], Fs[0], muv[0], rsd[0]);

    // ---- 2. layers 1..5 ----
    for (int i = 1; i < 6; ++i) {
        const int nthr = BATCH * Ts[i];
        term_layer<<<dim3((nthr + 255) / 256), 256, 0, stream>>>(
            h[i - 1], muv[i - 1], rsd[i - 1], gm[i - 1], bt[i - 1],
            W[i], bb[i], h[i], Ts[i], Ci[i], Fs[i - 1]);
        stats_kernel<<<dim3((Fs[i] + 63) / 64), 256, 0, stream>>>(h[i], Fs[i], muv[i], rsd[i]);
    }

    // ---- 3. final projection ----
    {
        const int nthr = BATCH * 768;
        final_proj<<<dim3((nthr + 255) / 256), 256, 0, stream>>>(
            h[5], muv[5], rsd[5], gm[5], bt[5], Wf, bfv, out);
    }
}
